// VS_DiT_Block_51814485459258
// MI455X (gfx1250) — compile-verified, matrix operand order checked by reading, not run on hardware
//
#include <hip/hip_runtime.h>
#include <hip/hip_bf16.h>
#include <cstdint>

typedef _Float16 v16h __attribute__((ext_vector_type(16)));
typedef _Float16 v8h  __attribute__((ext_vector_type(8)));
typedef float    v8f  __attribute__((ext_vector_type(8)));

#define DEV static __device__ __forceinline__

DEV v8h ld8(const _Float16* p) { return *(const v8h*)p; }
DEV v16h cat16(v8h lo, v8h hi) {
  return __builtin_shufflevector(lo, hi, 0,1,2,3,4,5,6,7,8,9,10,11,12,13,14,15);
}
DEV v8f wmma16(v16h a, v16h b, v8f c) {
  return __builtin_amdgcn_wmma_f32_16x16x32_f16(false, a, false, b, (short)0, c, false, false);
}

// ---- DPP16 width-16 reductions (VALU only, no ds_bpermute) ----
template <int CTRL>
DEV float dppf(float x) {
  int r = __builtin_amdgcn_update_dpp(0, __builtin_bit_cast(int, x), CTRL, 0xf, 0xf, true);
  return __builtin_bit_cast(float, r);
}
DEV float rmax16(float v) {
  v = fmaxf(v, dppf<0xB1>(v));   // quad_perm(1,0,3,2): xor1
  v = fmaxf(v, dppf<0x4E>(v));   // quad_perm(2,3,0,1): xor2
  v = fmaxf(v, dppf<0x141>(v));  // row_half_mirror  : xor4-equiv
  v = fmaxf(v, dppf<0x140>(v));  // row_mirror       : xor8-equiv
  return v;
}
DEV float rsum16(float v) {
  v += dppf<0xB1>(v);
  v += dppf<0x4E>(v);
  v += dppf<0x141>(v);
  v += dppf<0x140>(v);
  return v;
}

// ---------------- f32 -> f16 convert (plain) ----------------
__global__ void k_cvt_f16(const float* __restrict__ in, _Float16* __restrict__ out, int n) {
  for (int i = blockIdx.x * blockDim.x + threadIdx.x; i < n; i += gridDim.x * blockDim.x)
    out[i] = (_Float16)in[i];
}

// ---------------- f32 [K][N] -> f16 transposed [N][K] (tiled) ----------------
__global__ __launch_bounds__(256) void k_cvt_t(const float* __restrict__ in,
                                               _Float16* __restrict__ out, int K, int N) {
  __shared__ _Float16 t[32][33];
  const int n0 = blockIdx.x * 32, k0 = blockIdx.y * 32;
  const int tx = threadIdx.x & 31, ty = threadIdx.x >> 5;  // ty 0..7
#pragma unroll
  for (int r = 0; r < 32; r += 8)
    t[ty + r][tx] = (_Float16)in[(size_t)(k0 + ty + r) * N + n0 + tx];
  __syncthreads();
#pragma unroll
  for (int r = 0; r < 32; r += 8)
    out[(size_t)(n0 + ty + r) * K + k0 + tx] = t[tx][ty + r];
}

// ---------------- AdaLN modulation: mod[b, 6144] = silu(t_emb[b]) @ W + bias ----------------
__global__ __launch_bounds__(256) void k_mod(const float* __restrict__ t_emb,
                                             const float* __restrict__ W,
                                             const float* __restrict__ bias,
                                             float* __restrict__ mod) {
  __shared__ float st[1024];
  const int b = blockIdx.y;
  const int tid = threadIdx.x;
  const int c = blockIdx.x * 256 + tid;
  for (int i = tid; i < 1024; i += 256) {
    float t = t_emb[b * 1024 + i];
    st[i] = t / (1.0f + __expf(-t));
  }
  __syncthreads();
  float acc = bias[c];
#pragma unroll 4
  for (int k = 0; k < 1024; ++k) acc += st[k] * W[(size_t)k * 6144 + c];
  mod[b * 6144 + c] = acc;
}

// ---------------- LayerNorm (+ optional AdaLN scale/shift) -> f16 ----------------
__global__ __launch_bounds__(256) void k_ln(const float* __restrict__ x,
                                            const float* __restrict__ scb,
                                            const float* __restrict__ shb,
                                            _Float16* __restrict__ out) {
  __shared__ float red[34];
  const int row = blockIdx.x;
  const int b = row >> 10;  // rows per batch = 1024
  const int tid = threadIdx.x;
  float xv[4];
  float s = 0.f, q = 0.f;
#pragma unroll
  for (int p = 0; p < 4; ++p) {
    float v = x[(size_t)row * 1024 + tid + p * 256];
    xv[p] = v; s += v; q += v * v;
  }
  s = rsum16(s);
  q = rsum16(q);
  if ((tid & 15) == 0) { red[(tid >> 4) * 2] = s; red[(tid >> 4) * 2 + 1] = q; }
  __syncthreads();
  if (tid == 0) {
    float ts = 0.f, tq = 0.f;
    for (int w = 0; w < 16; ++w) { ts += red[2 * w]; tq += red[2 * w + 1]; }
    float mu = ts * (1.0f / 1024.0f);
    float var = tq * (1.0f / 1024.0f) - mu * mu;
    red[32] = mu;
    red[33] = rsqrtf(var + 1e-6f);
  }
  __syncthreads();
  float mu = red[32], rs = red[33];
#pragma unroll
  for (int p = 0; p < 4; ++p) {
    int col = tid + p * 256;
    float v = (xv[p] - mu) * rs;
    if (scb) v = v * (1.0f + scb[b * 6144 + col]) + shb[b * 6144 + col];
    out[(size_t)row * 1024 + col] = (_Float16)v;
  }
}

// ---------------- WMMA GEMM (LDS-free): C[M,N] = A_f16[M,K] @ Bt_f16[N,K]^T + bias ----------------
// Block 256 thr = 8 waves (4m x 2n), block tile 128x128, wave tile 32x64.
// EPI 0: store f16. EPI 1: tanh-GELU, store f16. EPI 2: out32 = resid + gate*(acc+bias).
template <int EPI>
__global__ __launch_bounds__(256) void k_gemm(const _Float16* __restrict__ A,
                                              const _Float16* __restrict__ Bt,
                                              const float* __restrict__ bias,
                                              int M, int Kd, int Nd,
                                              _Float16* __restrict__ out16,
                                              float* __restrict__ out32,
                                              const float* __restrict__ resid,
                                              const float* __restrict__ gate) {
  const int tid = threadIdx.x;
  const int lane = tid & 31, wid = tid >> 5;
  const int hf = lane >> 4, ln = lane & 15;
  const int wm = wid >> 1, wn = wid & 1;
  const int m_base = blockIdx.y * 128 + wm * 32;
  const int n_wave = blockIdx.x * 128 + wn * 64;

  v8f c[2][4] = {};
  for (int k0 = 0; k0 < Kd; k0 += 32) {
    v16h a[2], b[4];
#pragma unroll
    for (int i = 0; i < 2; ++i) {
      int row = m_base + i * 16 + ln;
      v8h lo = {}, hi = {};
      if (row < M) {
        const _Float16* ap = A + (size_t)row * Kd + k0 + 8 * hf;
        lo = ld8(ap); hi = ld8(ap + 16);
        if (i == 0 && k0 + 32 < Kd) __builtin_prefetch(ap + 32, 0, 1);
      }
      a[i] = cat16(lo, hi);
    }
#pragma unroll
    for (int j = 0; j < 4; ++j) {
      const _Float16* bp = Bt + (size_t)(n_wave + j * 16 + ln) * Kd + k0 + 8 * hf;
      b[j] = cat16(ld8(bp), ld8(bp + 16));
      if (j == 0 && k0 + 32 < Kd) __builtin_prefetch(bp + 32, 0, 1);
    }
#pragma unroll
    for (int i = 0; i < 2; ++i)
#pragma unroll
      for (int j = 0; j < 4; ++j) c[i][j] = wmma16(a[i], b[j], c[i][j]);
  }
#pragma unroll
  for (int i = 0; i < 2; ++i) {
#pragma unroll
    for (int r = 0; r < 8; ++r) {
      int row = m_base + i * 16 + r + 8 * hf;
      if (row >= M) continue;
#pragma unroll
      for (int j = 0; j < 4; ++j) {
        int col = n_wave + j * 16 + ln;
        float v = c[i][j][r] + bias[col];
        if (EPI == 1) {
          float u = v;
          v = 0.5f * u * (1.0f + tanhf(0.7978845608f * (u + 0.044715f * u * u * u)));
        }
        if (EPI <= 1) {
          out16[(size_t)row * Nd + col] = (_Float16)v;
        } else {
          float g = gate ? gate[(row >> 10) * 6144 + col] : 1.0f;
          out32[(size_t)row * Nd + col] = resid[(size_t)row * Nd + col] + g * v;
        }
      }
    }
  }
}

// ---------------- Flash attention: one wave per 16-query tile, 32-key chunks ----------------
// Q:[.., rstride] f16, K:[.., rstride] f16 (rows = keys), Vt:[B,H,HD,S_pad] f16 (pre-transposed),
// O:[.., rstride] f16.  Nq=1024, H=16 fixed.
__global__ __launch_bounds__(256) void k_attn(const _Float16* __restrict__ Q,
                                              unsigned long long q_bs, int q_rs,
                                              const _Float16* __restrict__ Km,
                                              unsigned long long k_bs, int k_rs,
                                              const _Float16* __restrict__ Vt,
                                              unsigned long long v_bs, int v_ss,
                                              _Float16* __restrict__ O,
                                              unsigned long long o_bs, int o_rs,
                                              int S, float scale) {
  __shared__ _Float16 pbuf[8 * 512];
  const int tid = threadIdx.x;
  const int wid = tid >> 5;
  const int lane = tid & 31;
  const int hf = lane >> 4;
  const int ln = lane & 15;
  _Float16* pw = pbuf + wid * 512;

  const int gid = blockIdx.x * 8 + wid;  // tile id over B*H*64
  const int qt = gid & 63;
  const int head = (gid >> 6) & 15;
  const int b = gid >> 10;

  const _Float16* Qb = Q + (unsigned long long)b * q_bs + head * 64;
  const _Float16* Kb = Km + (unsigned long long)b * k_bs + head * 64;
  const _Float16* Vb = Vt + (unsigned long long)b * v_bs + (unsigned long long)head * 64 * v_ss;
  _Float16* Ob = O + (unsigned long long)b * o_bs + head * 64;

  // Q fragments, pre-scaled by 1/sqrt(HD) (0.125, exact in f16)
  v16h qa[2];
  {
    const _Float16* qp = Qb + (unsigned long long)(qt * 16 + ln) * q_rs;
    const _Float16 hs = (_Float16)scale;
#pragma unroll
    for (int i = 0; i < 2; ++i) {
      v8h lo = ld8(qp + i * 32 + 8 * hf);
      v8h hi = ld8(qp + i * 32 + 16 + 8 * hf);
#pragma unroll
      for (int e = 0; e < 8; ++e) { lo[e] = lo[e] * hs; hi[e] = hi[e] * hs; }
      qa[i] = cat16(lo, hi);
    }
  }

  float Mr[8], Lr[8];
  v8f oacc[4] = {};
#pragma unroll
  for (int r = 0; r < 8; ++r) { Mr[r] = -3.0e38f; Lr[r] = 0.f; }

  for (int s0 = 0; s0 < S; s0 += 32) {
    v8f ct[2];
#pragma unroll
    for (int t = 0; t < 2; ++t) {  // scores for keys [s0+16t, s0+16t+16)
      int key = s0 + t * 16 + ln;
      bool ok = key < S;
      const _Float16* kp = Kb + (unsigned long long)key * k_rs;
      v8f c = {};
#pragma unroll
      for (int i = 0; i < 2; ++i) {
        v8h lo = {}, hi = {};
        if (ok) { lo = ld8(kp + i * 32 + 8 * hf); hi = ld8(kp + i * 32 + 16 + 8 * hf); }
        c = wmma16(qa[i], cat16(lo, hi), c);
      }
      if (!ok) {
#pragma unroll
        for (int r = 0; r < 8; ++r) c[r] = -1.0e30f;
      }
      ct[t] = c;
    }
    // running-max update (row r+8*hf lives on element r across 16 lanes)
    float scl[8];
#pragma unroll
    for (int r = 0; r < 8; ++r) {
      float v = rmax16(fmaxf(ct[0][r], ct[1][r]));
      float nm = fmaxf(Mr[r], v);
      scl[r] = __expf(Mr[r] - nm);
      Mr[r] = nm;
    }
#pragma unroll
    for (int t = 0; t < 2; ++t)
#pragma unroll
      for (int r = 0; r < 8; ++r) ct[t][r] = __expf(ct[t][r] - Mr[r]);
#pragma unroll
    for (int r = 0; r < 8; ++r) {
      float v = rsum16(ct[0][r] + ct[1][r]);
      Lr[r] = Lr[r] * scl[r] + v;
    }
#pragma unroll
    for (int r = 0; r < 8; ++r) {
      oacc[0][r] *= scl[r]; oacc[1][r] *= scl[r];
      oacc[2][r] *= scl[r]; oacc[3][r] *= scl[r];
    }
    // stage P (C-layout -> A-layout) via per-wave LDS scratch
#pragma unroll
    for (int t = 0; t < 2; ++t)
#pragma unroll
      for (int r = 0; r < 8; ++r)
        pw[(r + 8 * hf) * 32 + t * 16 + ln] = (_Float16)ct[t][r];
    asm volatile("s_wait_dscnt 0" ::: "memory");
    v16h pa;
    {
      const _Float16* pp = pw + ln * 32 + 8 * hf;
      pa = cat16(*(const v8h*)pp, *(const v8h*)(pp + 16));
    }
    // O += P @ V   (Vt rows are contiguous in s -> two b128 loads per fragment)
#pragma unroll
    for (int dt = 0; dt < 4; ++dt) {
      const _Float16* vp = Vb + (unsigned long long)(dt * 16 + ln) * v_ss + s0;
      oacc[dt] = wmma16(pa, cat16(ld8(vp + 8 * hf), ld8(vp + 16 + 8 * hf)), oacc[dt]);
    }
  }
#pragma unroll
  for (int r = 0; r < 8; ++r) {
    float inv = 1.0f / Lr[r];
    int row = qt * 16 + r + 8 * hf;
    _Float16* op = Ob + (unsigned long long)row * o_rs;
#pragma unroll
    for (int dt = 0; dt < 4; ++dt) op[dt * 16 + ln] = (_Float16)(oacc[dt][r] * inv);
  }
}

// ---------------- V transposes ----------------
__global__ void k_vt_self(const _Float16* __restrict__ qkv, _Float16* __restrict__ vt) {
  const int n = 4 * 16 * 64 * 1024;
  for (int i = blockIdx.x * blockDim.x + threadIdx.x; i < n; i += gridDim.x * blockDim.x) {
    int s = i & 1023, d = (i >> 10) & 63, h = (i >> 16) & 15, b = i >> 20;
    vt[i] = qkv[(size_t)(b * 1024 + s) * 3072 + 2048 + h * 64 + d];
  }
}
__global__ void k_vt_cross(const _Float16* __restrict__ kv, _Float16* __restrict__ vt) {
  const int n = 4 * 16 * 64 * 96;  // S padded 77 -> 96
  for (int i = blockIdx.x * blockDim.x + threadIdx.x; i < n; i += gridDim.x * blockDim.x) {
    int s = i % 96;
    int rest = i / 96;
    int d = rest & 63, h = (rest >> 6) & 15, b = rest >> 10;
    _Float16 v = (_Float16)0.f;
    if (s < 77) v = kv[(size_t)(b * 77 + s) * 2048 + 1024 + h * 64 + d];
    vt[i] = v;
  }
}

// ---------------- host ----------------
extern "C" void kernel_launch(void* const* d_in, const int* in_sizes, int n_in,
                              void* d_out, int out_size, void* d_ws, size_t ws_size,
                              hipStream_t stream) {
  const float* x       = (const float*)d_in[0];
  const float* t_emb   = (const float*)d_in[1];
  const float* ctx     = (const float*)d_in[2];
  const float* adaln_w = (const float*)d_in[3];
  const float* adaln_b = (const float*)d_in[4];
  const float* qkv_w   = (const float*)d_in[5];
  const float* qkv_b   = (const float*)d_in[6];
  const float* saw     = (const float*)d_in[7];
  const float* sab     = (const float*)d_in[8];
  const float* qw      = (const float*)d_in[9];
  const float* qb      = (const float*)d_in[10];
  const float* kvw     = (const float*)d_in[11];
  const float* kvb     = (const float*)d_in[12];
  const float* caw     = (const float*)d_in[13];
  const float* cab     = (const float*)d_in[14];
  const float* f1w     = (const float*)d_in[15];
  const float* f1b     = (const float*)d_in[16];
  const float* f2w     = (const float*)d_in[17];
  const float* f2b     = (const float*)d_in[18];
  float* out = (float*)d_out;

  char* p = (char*)d_ws;
  auto alloc = [&](size_t bytes) -> char* {
    char* r = p;
    p += (bytes + 255) & ~(size_t)255;
    return r;
  };
  float*    modb    = (float*)alloc((size_t)4 * 6144 * 4);
  _Float16* qkvw16  = (_Float16*)alloc((size_t)1024 * 3072 * 2);  // transposed [N][K]
  _Float16* saw16   = (_Float16*)alloc((size_t)1024 * 1024 * 2);
  _Float16* qw16    = (_Float16*)alloc((size_t)1024 * 1024 * 2);
  _Float16* kvw16   = (_Float16*)alloc((size_t)1024 * 2048 * 2);
  _Float16* caw16   = (_Float16*)alloc((size_t)1024 * 1024 * 2);
  _Float16* f1w16   = (_Float16*)alloc((size_t)1024 * 4096 * 2);
  _Float16* f2w16   = (_Float16*)alloc((size_t)4096 * 1024 * 2);
  _Float16* ctx16   = (_Float16*)alloc((size_t)4 * 77 * 1024 * 2);
  _Float16* lnout   = (_Float16*)alloc((size_t)4096 * 1024 * 2);   // reused 3x
  _Float16* qkv16a  = (_Float16*)alloc((size_t)4096 * 3072 * 2);   // later reused as q16
  _Float16* vt_sa   = (_Float16*)alloc((size_t)4 * 16 * 64 * 1024 * 2);
  _Float16* attn16  = (_Float16*)alloc((size_t)4096 * 1024 * 2);   // reused 2x
  float*    x1      = (float*)alloc((size_t)4096 * 1024 * 4);
  _Float16* kv16    = (_Float16*)alloc((size_t)308 * 2048 * 2);
  _Float16* vt_ca   = (_Float16*)alloc((size_t)4 * 16 * 64 * 96 * 2);
  float*    x2      = (float*)alloc((size_t)4096 * 1024 * 4);
  _Float16* h16     = (_Float16*)alloc((size_t)4096 * 4096 * 2);
  _Float16* q16     = qkv16a;  // alias: qkv activations dead after self-attention

  auto cvt_t = [&](const float* src, _Float16* dst, int K, int N) {
    k_cvt_t<<<dim3(N / 32, K / 32), 256, 0, stream>>>(src, dst, K, N);
  };
  // 1) weight conversion to transposed f16 [N][K]; context stays row-major f16
  cvt_t(qkv_w, qkvw16, 1024, 3072);
  cvt_t(saw, saw16, 1024, 1024);
  cvt_t(qw, qw16, 1024, 1024);
  cvt_t(kvw, kvw16, 1024, 2048);
  cvt_t(caw, caw16, 1024, 1024);
  cvt_t(f1w, f1w16, 1024, 4096);
  cvt_t(f2w, f2w16, 4096, 1024);
  k_cvt_f16<<<(4 * 77 * 1024 + 2047) / 2048, 256, 0, stream>>>(ctx, ctx16, 4 * 77 * 1024);

  // 2) AdaLN modulation
  k_mod<<<dim3(24, 4), 256, 0, stream>>>(t_emb, adaln_w, adaln_b, modb);

  // ---- self-attention branch ----
  k_ln<<<4096, 256, 0, stream>>>(x, modb + 1024, modb + 0, lnout);  // sc_sa, sh_sa
  k_gemm<0><<<dim3(24, 32), 256, 0, stream>>>(lnout, qkvw16, qkv_b, 4096, 1024, 3072,
                                              qkv16a, nullptr, nullptr, nullptr);
  k_vt_self<<<4096, 256, 0, stream>>>(qkv16a, vt_sa);
  k_attn<<<512, 256, 0, stream>>>(qkv16a, 1024ULL * 3072, 3072,
                                  qkv16a + 1024, 1024ULL * 3072, 3072,
                                  vt_sa, 16ULL * 64 * 1024, 1024,
                                  attn16, 1024ULL * 1024, 1024, 1024, 0.125f);
  k_gemm<2><<<dim3(8, 32), 256, 0, stream>>>(attn16, saw16, sab, 4096, 1024, 1024,
                                             nullptr, x1, x, modb + 2 * 1024);  // g_sa

  // ---- cross-attention branch ----
  k_ln<<<4096, 256, 0, stream>>>(x1, nullptr, nullptr, lnout);
  k_gemm<0><<<dim3(8, 32), 256, 0, stream>>>(lnout, qw16, qb, 4096, 1024, 1024,
                                             q16, nullptr, nullptr, nullptr);
  k_gemm<0><<<dim3(16, 3), 256, 0, stream>>>(ctx16, kvw16, kvb, 308, 1024, 2048,
                                             kv16, nullptr, nullptr, nullptr);
  k_vt_cross<<<512, 256, 0, stream>>>(kv16, vt_ca);
  k_attn<<<512, 256, 0, stream>>>(q16, 1024ULL * 1024, 1024,
                                  kv16, 77ULL * 2048, 2048,
                                  vt_ca, 16ULL * 64 * 96, 96,
                                  attn16, 1024ULL * 1024, 1024, 77, 0.125f);
  k_gemm<2><<<dim3(8, 32), 256, 0, stream>>>(attn16, caw16, cab, 4096, 1024, 1024,
                                             nullptr, x2, x1, nullptr);  // gate=1

  // ---- MLP branch ----
  k_ln<<<4096, 256, 0, stream>>>(x2, modb + 4 * 1024, modb + 3 * 1024, lnout);  // sc_ff, sh_ff
  k_gemm<1><<<dim3(32, 32), 256, 0, stream>>>(lnout, f1w16, f1b, 4096, 1024, 4096,
                                              h16, nullptr, nullptr, nullptr);
  k_gemm<2><<<dim3(8, 32), 256, 0, stream>>>(h16, f2w16, f2b, 4096, 4096, 1024,
                                             nullptr, out, x2, modb + 5 * 1024);  // g_ff
}
